// DualTower_81415400063072
// MI455X (gfx1250) — matrix operand order repaired and checked by reading, not compiled, not run on hardware
//
#include <hip/hip_runtime.h>
#include <hip/hip_bf16.h>
#include <hip/hip_fp16.h>

// ---------------------------------------------------------------------------
// MI455X / gfx1250: all GEMMs + attention via v_wmma_f32_16x16x32_f16
// (wave32, f32 accumulate), async global->LDS staging (ASYNCcnt path).
// ---------------------------------------------------------------------------

typedef __attribute__((ext_vector_type(16))) _Float16 v16h;
typedef __attribute__((ext_vector_type(8)))  _Float16 v8h;
typedef __attribute__((ext_vector_type(4)))  _Float16 v4h;
typedef __attribute__((ext_vector_type(8)))  float    v8f;
typedef __attribute__((ext_vector_type(4)))  int      v4i;

#define B_    512
#define S_    200
#define D_    512
#define H_    8
#define DK_   64
#define FF_   1024
#define EMB_  128
#define HID_  1024
#define FIN_  64
#define BS_   (B_*S_)
#define UDIM_ 1152   // D + 5*EMB

#define AS1 __attribute__((address_space(1)))
#define AS3 __attribute__((address_space(3)))

#if __has_builtin(__builtin_amdgcn_global_load_async_to_lds_b128)
#define HAS_ASYNC_LDS 1
#else
#define HAS_ASYNC_LDS 0
#endif

// async copy of 16B global -> LDS (falls back to sync copy through VGPRs)
__device__ __forceinline__ void async_cp16(const _Float16* gp, _Float16* lp) {
#if HAS_ASYNC_LDS
  __builtin_amdgcn_global_load_async_to_lds_b128(
      (AS1 v4i*)(const void*)gp, (AS3 v4i*)(void*)lp, 0, 0);
#else
  *(v8h*)lp = *(const v8h*)gp;
#endif
}

__device__ __forceinline__ void wait_async0() {
#if HAS_ASYNC_LDS
#if __has_builtin(__builtin_amdgcn_s_wait_asynccnt)
  __builtin_amdgcn_s_wait_asynccnt(0);
#else
  asm volatile("s_wait_asynccnt 0x0" ::: "memory");
#endif
#endif
}

// fragment helper: build v16h from two 8-half (16B) chunks
__device__ __forceinline__ v16h frag16(const _Float16* p0, const _Float16* p1) {
  v8h lo = *(const v8h*)p0;
  v8h hi = *(const v8h*)p1;
  v16h r;
#pragma unroll
  for (int e = 0; e < 8; ++e) { r[e] = lo[e]; r[8+e] = hi[e]; }
  return r;
}

// ===========================================================================
// Generic tiled WMMA GEMM:  C(MxN) = act( A(MxK,f16,row) * W(NxK,f32,row)^T + bias )
// Block tile 128x64, K-step 32, double-buffered LDS, async A staging.
// 8 waves -> each wave a 32x32 sub-tile (4 wmma per K-step).
// ===========================================================================
#define BM 128
#define BN 64
#define BK 32
#define LDA 40   // halfs; 80B rows -> conflict-free 16B frag reads
#define LDB 40

#define ACT_NONE      0
#define ACT_CLIP      1
#define ACT_CLIP_RELU 2
#define ACT_RELU      3

template<int ACT, bool OUTH>
__global__ __launch_bounds__(256) void gemm_kernel(
    const _Float16* __restrict__ A, const float* __restrict__ W,
    const float* __restrict__ bias, void* __restrict__ Cout,
    int M, int N, int K, float clipv)
{
  __shared__ _Float16 Ash[2][BM*LDA];
  __shared__ _Float16 Bsh[2][BN*LDB];

  const int tid  = threadIdx.x;
  const int lane = tid & 31;
  const int wid  = tid >> 5;
  const int wm   = wid & 3;     // 0..3 (M dir)
  const int wn   = wid >> 2;    // 0..1 (N dir)
  const int lg   = lane >> 4;   // half-wave group
  const int lr   = lane & 15;
  const int mTile = blockIdx.y * BM;
  const int nTile = blockIdx.x * BN;

  v8f c[2][2];
#pragma unroll
  for (int i = 0; i < 2; ++i)
#pragma unroll
    for (int j = 0; j < 2; ++j) c[i][j] = {};

  const int arow = tid >> 1;              // A staging: 2 threads/row (2x16B)
  const int acol = (tid & 1) * 16;
  const int brow = tid >> 2;              // B staging: 4 threads/row (8 f32)
  const int bcol = (tid & 3) * 8;

  // stage K-tile `kk` into LDS buffer `buf`
  auto stage = [&](int kk, int buf) {
    // A tile (f16, 128x32): async global->LDS
    const _Float16* gp = A + (size_t)(mTile + arow) * K + kk + acol;
    _Float16* lp = &Ash[buf][arow*LDA + acol];
    async_cp16(gp,     lp);
    async_cp16(gp + 8, lp + 8);
    // B tile (f32 -> f16, 64x32): through VGPRs (needs conversion)
    const float* wp = W + (size_t)(nTile + brow) * K + kk + bcol;
    float4 w0 = *(const float4*)(wp);
    float4 w1 = *(const float4*)(wp + 4);
    v8h bh;
    bh[0] = (_Float16)w0.x; bh[1] = (_Float16)w0.y;
    bh[2] = (_Float16)w0.z; bh[3] = (_Float16)w0.w;
    bh[4] = (_Float16)w1.x; bh[5] = (_Float16)w1.y;
    bh[6] = (_Float16)w1.z; bh[7] = (_Float16)w1.w;
    *(v8h*)(&Bsh[buf][brow*LDB + bcol]) = bh;
  };

  const int nsteps = K / BK;
  stage(0, 0);
  wait_async0();
  __syncthreads();

  for (int t = 0; t < nsteps; ++t) {
    const int cur = t & 1;
    if (t + 1 < nsteps) stage((t + 1) * BK, cur ^ 1);   // prefetch next tile

    // fragments (ISA 7.12.2 layouts)
    v16h a[2], b[2];
#pragma unroll
    for (int i = 0; i < 2; ++i) {
      const _Float16* p = &Ash[cur][(wm*32 + i*16 + lr)*LDA + lg*8];
      a[i] = frag16(p, p + 16);
    }
#pragma unroll
    for (int j = 0; j < 2; ++j) {
      const _Float16* p = &Bsh[cur][(wn*32 + j*16 + lr)*LDB + lg*8];
      b[j] = frag16(p, p + 16);
    }
#pragma unroll
    for (int i = 0; i < 2; ++i)
#pragma unroll
      for (int j = 0; j < 2; ++j)
        c[i][j] = __builtin_amdgcn_wmma_f32_16x16x32_f16(
            false, a[i], false, b[j], (short)0, c[i][j], false, false);

    wait_async0();       // next-tile async writes landed in LDS
    __syncthreads();     // all waves done reading `cur`, next tile visible
  }

  // ---- epilogue ----
#pragma unroll
  for (int i = 0; i < 2; ++i) {
#pragma unroll
    for (int j = 0; j < 2; ++j) {
      const int col = nTile + wn*32 + j*16 + lr;
      const float bv = bias ? bias[col] : 0.f;
#pragma unroll
      for (int r = 0; r < 8; ++r) {
        const int row = mTile + wm*32 + i*16 + lg*8 + r;
        float v = c[i][j][r] + bv;
        if (ACT == ACT_CLIP)       v = fminf(fmaxf(v, -clipv), clipv);
        else if (ACT == ACT_CLIP_RELU) v = fminf(fmaxf(v, 0.f), clipv); // clip then relu
        else if (ACT == ACT_RELU)  v = fmaxf(v, 0.f);
        if (OUTH) ((_Float16*)Cout)[(size_t)row*N + col] = (_Float16)v;
        else      ((float*)Cout)[(size_t)row*N + col]    = v;
      }
    }
  }
}

// ===========================================================================
// Embedding gather: x = clip(emb[seq]*0.5, +-1) -> f32 residual + f16 operand
// ===========================================================================
__global__ __launch_bounds__(256) void embed_kernel(
    const int* __restrict__ seq, const float* __restrict__ emb,
    float* __restrict__ xf, _Float16* __restrict__ xh)
{
  const size_t idx = (size_t)blockIdx.x * 256 + threadIdx.x;   // per float4
  const size_t tok = idx / (D_/4);
  const int    c4  = (int)(idx % (D_/4));
  const int    it  = seq[tok];
  float4 e = ((const float4*)(emb + (size_t)it * D_))[c4];
  float4 o;
  o.x = fminf(fmaxf(e.x*0.5f, -1.f), 1.f);
  o.y = fminf(fmaxf(e.y*0.5f, -1.f), 1.f);
  o.z = fminf(fmaxf(e.z*0.5f, -1.f), 1.f);
  o.w = fminf(fmaxf(e.w*0.5f, -1.f), 1.f);
  ((float4*)xf)[idx] = o;
  v4h h; h[0]=(_Float16)o.x; h[1]=(_Float16)o.y; h[2]=(_Float16)o.z; h[3]=(_Float16)o.w;
  *(v4h*)(xh + idx*4) = h;
}

// ===========================================================================
// Fused attention, one workgroup per (b,h). K/V resident in LDS (320KB/WGP).
// S padded 200 -> 224. 8 waves, each owns q-tiles of 16 rows.
// ===========================================================================
#define SP  224
#define KST 72    // Ks row stride (halfs): conflict-free 16B frag reads
#define VST 232   // Vt row stride
#define PST 232   // P  row stride
#define NQT 13    // ceil(200/16)
#define NKT 14    // SP/16
#define NSC 7     // SP/32

__global__ __launch_bounds__(256) void attn_kernel(
    const _Float16* __restrict__ qkv, const int* __restrict__ seq,
    _Float16* __restrict__ ctx)
{
  __shared__ _Float16 Ks[SP*KST];        // K, row-major in dk
  __shared__ _Float16 Vt[DK_*VST];       // V transposed (dk-major)
  __shared__ _Float16 Pb[8*16*PST];      // per-wave softmax probs
  __shared__ float    maskAdd[SP];

  const int b    = blockIdx.x / H_;
  const int h    = blockIdx.x % H_;
  const int tid  = threadIdx.x;
  const int lane = tid & 31;
  const int wid  = tid >> 5;
  const int lg   = lane >> 4;
  const int lr   = lane & 15;
  const size_t tokBase = (size_t)b * S_;

  // ---- stage K (async, row-major) and V (transposed) from qkv [BS,1536] ----
  for (int idx = tid; idx < SP*8; idx += 256) {
    const int row = idx >> 3;
    const int col = (idx & 7) * 8;
    if (row < S_) {
      const _Float16* base = qkv + (tokBase + row) * (3*D_) + h*DK_ + col;
      async_cp16(base + D_, &Ks[row*KST + col]);       // K section
      v8h vv = *(const v8h*)(base + 2*D_);             // V section
#pragma unroll
      for (int e = 0; e < 8; ++e) Vt[(col+e)*VST + row] = vv[e];
    } else {
      v8h z = {};
      *(v8h*)(&Ks[row*KST + col]) = z;
#pragma unroll
      for (int e = 0; e < 8; ++e) Vt[(col+e)*VST + row] = (_Float16)0.f;
    }
  }
  for (int s = tid; s < SP; s += 256)
    maskAdd[s] = (s < S_ && seq[tokBase + s] != 0) ? 0.f : -1e9f;
  wait_async0();
  __syncthreads();

  _Float16* Pw = &Pb[wid * 16 * PST];

  for (int qt = wid; qt < NQT; qt += 8) {
    const int qBase = qt * 16;

    // Q fragments straight from global (reused across 14 key tiles)
    int qr = qBase + lr; if (qr > S_-1) qr = S_-1;
    const _Float16* qp = qkv + (tokBase + qr) * (3*D_) + h*DK_;
    v16h qa[2];
#pragma unroll
    for (int cc = 0; cc < 2; ++cc)
      qa[cc] = frag16(qp + cc*32 + lg*8, qp + cc*32 + 16 + lg*8);

    // ---- scores: S = clip(2.5 * Q K^T, +-3) + mask ----
    v8f sc[NKT];
#pragma unroll
    for (int kt = 0; kt < NKT; ++kt) {
      v8f acc = {};
#pragma unroll
      for (int cc = 0; cc < 2; ++cc) {
        const _Float16* p = &Ks[(kt*16 + lr)*KST + cc*32 + lg*8];
        v16h bk = frag16(p, p + 16);
        acc = __builtin_amdgcn_wmma_f32_16x16x32_f16(
            false, qa[cc], false, bk, (short)0, acc, false, false);
      }
      const float madd = maskAdd[kt*16 + lr];
#pragma unroll
      for (int r = 0; r < 8; ++r) {
        float s = acc[r] * 2.5f;               // 1/(sqrt(64)*0.05)
        acc[r] = fminf(fmaxf(s, -3.f), 3.f) + madd;
      }
      sc[kt] = acc;
    }

    // ---- row softmax (rows live across the 16 lanes of a half-group) ----
    float inv[8];
#pragma unroll
    for (int r = 0; r < 8; ++r) {
      float m = -1e30f;
#pragma unroll
      for (int kt = 0; kt < NKT; ++kt) m = fmaxf(m, sc[kt][r]);
      m = fmaxf(m, __shfl_xor(m, 1, 32));
      m = fmaxf(m, __shfl_xor(m, 2, 32));
      m = fmaxf(m, __shfl_xor(m, 4, 32));
      m = fmaxf(m, __shfl_xor(m, 8, 32));
      float ssum = 0.f;
#pragma unroll
      for (int kt = 0; kt < NKT; ++kt) {
        const float e = __expf(sc[kt][r] - m);
        ssum += e;
        Pw[(lg*8 + r)*PST + kt*16 + lr] = (_Float16)e;  // unnormalized P
      }
      ssum += __shfl_xor(ssum, 1, 32);
      ssum += __shfl_xor(ssum, 2, 32);
      ssum += __shfl_xor(ssum, 4, 32);
      ssum += __shfl_xor(ssum, 8, 32);
      inv[r] = 1.f / ssum;                     // fold 1/sum into epilogue
    }
    // same-wave LDS RAW on Pw: drain DS counter before fragment reads
    asm volatile("s_wait_dscnt 0x0" ::: "memory");

    // ---- ctx = P @ V  (16x224 x 224x64) ----
    v8f cv[4];
#pragma unroll
    for (int j = 0; j < 4; ++j) cv[j] = {};
#pragma unroll
    for (int st = 0; st < NSC; ++st) {
      const _Float16* pp = &Pw[lr*PST + st*32 + lg*8];
      v16h pa = frag16(pp, pp + 16);
#pragma unroll
      for (int j = 0; j < 4; ++j) {
        const _Float16* vp = &Vt[(j*16 + lr)*VST + st*32 + lg*8];
        v16h bv = frag16(vp, vp + 16);
        cv[j] = __builtin_amdgcn_wmma_f32_16x16x32_f16(
            false, pa, false, bv, (short)0, cv[j], false, false);
      }
    }
    // ---- store ctx (already in (b,s,h,dk)->[BS,D] layout) ----
#pragma unroll
    for (int j = 0; j < 4; ++j)
#pragma unroll
      for (int r = 0; r < 8; ++r) {
        const int q = qBase + lg*8 + r;
        if (q < S_)
          ctx[(tokBase + q)*D_ + h*DK_ + j*16 + lr] = (_Float16)(cv[j][r] * inv[r]);
      }
  }
}

// ===========================================================================
// Fused residual + double LayerNorm:  out = LN(x + LN(x + y))
// One block per token row (D=512), 256 threads x 2 elements.
// ===========================================================================
__device__ __forceinline__ void block_reduce2(float& a, float& b, float* sbuf) {
#pragma unroll
  for (int m = 16; m >= 1; m >>= 1) { a += __shfl_xor(a, m, 32); b += __shfl_xor(b, m, 32); }
  const int wid = threadIdx.x >> 5, lane = threadIdx.x & 31;
  __syncthreads();
  if (lane == 0) { sbuf[wid] = a; sbuf[8 + wid] = b; }
  __syncthreads();
  float ta = 0.f, tb = 0.f;
#pragma unroll
  for (int i = 0; i < 8; ++i) { ta += sbuf[i]; tb += sbuf[8 + i]; }
  a = ta; b = tb;
}

template<bool FINAL_CLIP, bool WRITE_H>
__global__ __launch_bounds__(256) void add_ln2_kernel(
    const float* __restrict__ X, const float* __restrict__ Y,
    const float* __restrict__ g, const float* __restrict__ bt,
    float* __restrict__ outF, _Float16* __restrict__ outH)
{
  __shared__ float sbuf[16];
  const size_t row = blockIdx.x;
  const int t = threadIdx.x;
  const float* xr = X + row * D_;
  const float* yr = Y + row * D_;
  const float x0 = xr[t], x1 = xr[t + 256];
  float t0 = x0 + yr[t], t1 = x1 + yr[t + 256];

  float s = t0 + t1, q = t0*t0 + t1*t1;
  block_reduce2(s, q, sbuf);
  float mean = s * (1.f / D_);
  float var  = q * (1.f / D_) - mean * mean;
  float rstd = rsqrtf(var + 1e-6f);
  const float sa0 = (t0 - mean) * rstd * g[t]       + bt[t];
  const float sa1 = (t1 - mean) * rstd * g[t + 256] + bt[t + 256];

  t0 = x0 + sa0; t1 = x1 + sa1;
  s = t0 + t1; q = t0*t0 + t1*t1;
  block_reduce2(s, q, sbuf);
  mean = s * (1.f / D_);
  var  = q * (1.f / D_) - mean * mean;
  rstd = rsqrtf(var + 1e-6f);
  float o0 = (t0 - mean) * rstd * g[t]       + bt[t];
  float o1 = (t1 - mean) * rstd * g[t + 256] + bt[t + 256];
  if (FINAL_CLIP) { o0 = fminf(fmaxf(o0, -5.f), 5.f); o1 = fminf(fmaxf(o1, -5.f), 5.f); }
  outF[row*D_ + t]       = o0;
  outF[row*D_ + t + 256] = o1;
  if (WRITE_H) {
    outH[row*D_ + t]       = (_Float16)o0;
    outH[row*D_ + t + 256] = (_Float16)o1;
  }
}

// ===========================================================================
// Masked mean pooling over S -> u[:, 0:512] (f16)
// ===========================================================================
__global__ __launch_bounds__(256) void seqrep_kernel(
    const float* __restrict__ seqout, const int* __restrict__ seq,
    _Float16* __restrict__ uh)
{
  const int b = blockIdx.x, t = threadIdx.x;
  float a0 = 0.f, a1 = 0.f, cnt = 0.f;
  for (int s = 0; s < S_; ++s) {
    const float m = (seq[(size_t)b*S_ + s] != 0) ? 1.f : 0.f;
    cnt += m;
    const float* r = seqout + ((size_t)b*S_ + s) * D_;
    a0 += r[t] * m; a1 += r[t + 256] * m;
  }
  const float inv = 1.f / (cnt + 1e-8f);
  a0 = fminf(fmaxf(a0 * inv, -5.f), 5.f);
  a1 = fminf(fmaxf(a1 * inv, -5.f), 5.f);
  uh[(size_t)b*UDIM_ + t]       = (_Float16)a0;
  uh[(size_t)b*UDIM_ + t + 256] = (_Float16)a1;
}

// ===========================================================================
// User-feature concat -> u[:, 512:1152] (f16)
// ===========================================================================
__global__ __launch_bounds__(128) void feat_kernel(
    const float* __restrict__ ctr_s, const float* __restrict__ ti_s,
    const int* __restrict__ age, const int* __restrict__ gen, const int* __restrict__ cms,
    const float* __restrict__ age_tab, const float* __restrict__ gen_tab,
    const float* __restrict__ cms_tab,
    const float* __restrict__ ctr_w, const float* __restrict__ ctr_b,
    const float* __restrict__ ti_w,  const float* __restrict__ ti_b,
    _Float16* __restrict__ uh)
{
  const int b = blockIdx.x, t = threadIdx.x;   // t < 128
  const size_t base = (size_t)b * UDIM_;
  uh[base +  512 + t] = (_Float16)(ctr_s[b] * ctr_w[t] + ctr_b[t]);
  uh[base +  640 + t] = (_Float16)(ti_s[b]  * ti_w[t]  + ti_b[t]);
  uh[base +  768 + t] = (_Float16)age_tab[(size_t)age[b] * EMB_ + t];
  uh[base +  896 + t] = (_Float16)gen_tab[(size_t)gen[b] * EMB_ + t];
  uh[base + 1024 + t] = (_Float16)cms_tab[(size_t)cms[b] * EMB_ + t];
}

// ===========================================================================
// Launch
// ===========================================================================
extern "C" void kernel_launch(void* const* d_in, const int* in_sizes, int n_in,
                              void* d_out, int out_size, void* d_ws, size_t ws_size,
                              hipStream_t stream) {
  (void)in_sizes; (void)n_in; (void)out_size; (void)ws_size;

  const int*   item_seq  = (const int*)  d_in[0];
  const float* u_ctr     = (const float*)d_in[1];
  const float* u_ti      = (const float*)d_in[2];
  const int*   age_price = (const int*)  d_in[3];
  const int*   gender    = (const int*)  d_in[4];
  const int*   cmsg      = (const int*)  d_in[5];
  const float* emb       = (const float*)d_in[6];
  const float* in_proj_w = (const float*)d_in[7];
  const float* out_proj_w= (const float*)d_in[8];
  const float* out_proj_b= (const float*)d_in[9];
  const float* ln1_g     = (const float*)d_in[10];
  const float* ln1_b     = (const float*)d_in[11];
  const float* ln2_g     = (const float*)d_in[12];
  const float* ln2_b     = (const float*)d_in[13];
  const float* lin1_w    = (const float*)d_in[14];
  const float* lin1_b    = (const float*)d_in[15];
  const float* lin2_w    = (const float*)d_in[16];
  const float* lin2_b    = (const float*)d_in[17];
  const float* age_tab   = (const float*)d_in[18];
  const float* gen_tab   = (const float*)d_in[19];
  const float* cms_tab   = (const float*)d_in[20];
  const float* ctr_w     = (const float*)d_in[21];
  const float* ctr_b     = (const float*)d_in[22];
  const float* ti_w      = (const float*)d_in[23];
  const float* ti_b      = (const float*)d_in[24];
  const float* mlp1_w    = (const float*)d_in[25];
  const float* mlp1_b    = (const float*)d_in[26];
  const float* mlp2_w    = (const float*)d_in[27];
  const float* mlp2_b    = (const float*)d_in[28];

  char* ws = (char*)d_ws;
  size_t off = 0;
  auto take = [&](size_t bytes) -> char* {
    char* p = ws + off;
    off = (off + bytes + 255) & ~(size_t)255;
    return p;
  };
  float*     xf32 = (float*)    take((size_t)BS_*D_*4);     // embeddings; later seq_out
  _Float16*  xh   = (_Float16*) take((size_t)BS_*D_*2);
  _Float16*  qkvh = (_Float16*) take((size_t)BS_*3*D_*2);
  _Float16*  ctxh = (_Float16*) take((size_t)BS_*D_*2);
  float*     tmpf = (float*)    take((size_t)BS_*D_*4);     // GEMM f32 outs
  float*     xa   = (float*)    take((size_t)BS_*D_*4);     // post-attn residual
  _Float16*  xah  = (_Float16*) take((size_t)BS_*D_*2);
  _Float16*  hh   = (_Float16*) take((size_t)BS_*FF_*2);
  _Float16*  uh   = (_Float16*) take((size_t)B_*UDIM_*2);
  _Float16*  h1h  = (_Float16*) take((size_t)B_*HID_*2);

  // 1) embedding gather
  embed_kernel<<<(BS_*(size_t)D_/4)/256, 256, 0, stream>>>(item_seq, emb, xf32, xh);

  // 2) QKV projection (no bias), clip +-1, f16 out
  gemm_kernel<ACT_CLIP, true><<<dim3(3*D_/BN, BS_/BM), 256, 0, stream>>>(
      xh, in_proj_w, nullptr, qkvh, BS_, 3*D_, D_, 1.0f);

  // 3) fused attention
  attn_kernel<<<B_*H_, 256, 0, stream>>>(qkvh, item_seq, ctxh);

  // 4) out-proj, +bias, clip +-3, f32 out
  gemm_kernel<ACT_CLIP, false><<<dim3(D_/BN, BS_/BM), 256, 0, stream>>>(
      ctxh, out_proj_w, out_proj_b, tmpf, BS_, D_, D_, 3.0f);

  // 5) x = LN(x + LN(x + x2)) with ln1
  add_ln2_kernel<false, true><<<BS_, 256, 0, stream>>>(xf32, tmpf, ln1_g, ln1_b, xa, xah);

  // 6) FFN lin1: clip +-2 then relu, f16 out
  gemm_kernel<ACT_CLIP_RELU, true><<<dim3(FF_/BN, BS_/BM), 256, 0, stream>>>(
      xah, lin1_w, lin1_b, hh, BS_, FF_, D_, 2.0f);

  // 7) FFN lin2: clip +-2, f32 out
  gemm_kernel<ACT_CLIP, false><<<dim3(D_/BN, BS_/BM), 256, 0, stream>>>(
      hh, lin2_w, lin2_b, tmpf, BS_, D_, FF_, 2.0f);

  // 8) x = LN(x + LN(x + f2)) with ln2, clip +-5 -> seq_out (reuse xf32)
  add_ln2_kernel<true, false><<<BS_, 256, 0, stream>>>(xa, tmpf, ln2_g, ln2_b, xf32, nullptr);

  // 9) masked mean pooling -> u[:, :512]
  seqrep_kernel<<<B_, 256, 0, stream>>>(xf32, item_seq, uh);

  // 10) user features -> u[:, 512:]
  feat_kernel<<<B_, EMB_, 0, stream>>>(u_ctr, u_ti, age_price, gender, cmsg,
                                       age_tab, gen_tab, cms_tab,
                                       ctr_w, ctr_b, ti_w, ti_b, uh);

  // 11) MLP1: relu, f16 out
  gemm_kernel<ACT_RELU, true><<<dim3(HID_/BN, B_/BM), 256, 0, stream>>>(
      uh, mlp1_w, mlp1_b, h1h, B_, HID_, UDIM_, 0.f);

  // 12) MLP2 -> d_out (f32)
  gemm_kernel<ACT_NONE, false><<<dim3(FIN_/BN, B_/BM), 256, 0, stream>>>(
      h1h, mlp2_w, mlp2_b, (float*)d_out, B_, FIN_, HID_, 0.f);
}
